// MultiHeadAttention_84404697301533
// MI455X (gfx1250) — matrix operand order repaired and checked by reading, not compiled, not run on hardware
//
#include <hip/hip_runtime.h>
#include <hip/hip_bf16.h>

// ---------------------------------------------------------------------------
// MI455X (gfx1250) multi-head attention.
//   * all matmuls:  v_wmma_f32_16x16x32_bf16 (wave32, f32 accum)
//   * GEMM A-slab:  TENSOR_LOAD_TO_LDS (TDM descriptor, TENSORcnt)
//   * GEMM B-slab:  GLOBAL_LOAD_ASYNC_TO_LDS_B128 (ASYNCcnt)
//   * double-buffered LDS slabs, padded rows (80 B) for bank spread
// Fragment layouts follow CDNA5 ISA:
//   A 16x32 bf16 : lane l<16 -> row M=l, K=(0..7,16..23); lane l+16 -> K=(8..15,24..31)
//   B 32x16 bf16 : lane n<16 -> col N=n, K=(0..7,16..23); lane n+16 -> K=(8..15,24..31)
//   C/D 16x16 f32: VGPR r -> M=r (lanes 0-15, N=lane) / M=8+r (lanes 16-31, N=lane-16)
// ---------------------------------------------------------------------------

typedef __bf16 bf16_t;
typedef __attribute__((ext_vector_type(16))) __bf16        v16bf;
typedef __attribute__((ext_vector_type(8)))  float         v8f;
typedef __attribute__((ext_vector_type(4)))  unsigned int  u32x4;
typedef __attribute__((ext_vector_type(8)))  int           i32x8;
typedef __attribute__((ext_vector_type(4)))  int           i32x4;

union Frag { v16bf v; u32x4 q[2]; };

#define WMMA_BF16(a, b, c) \
  __builtin_amdgcn_wmma_f32_16x16x32_bf16(false, (a), false, (b), (short)0, (c), false, false)

constexpr int Bb   = 4;
constexpr int Ss   = 2048;
constexpr int DIMc = 1024;
constexpr int Hh   = 16;
constexpr int Dh   = 64;
constexpr int Mrows = Bb * Ss;   // 8192

__device__ inline v8f vzero8() {
  v8f z;
#pragma unroll
  for (int i = 0; i < 8; ++i) z[i] = 0.0f;
  return z;
}

// ---- CDNA5 async global->LDS copy (ASYNCcnt) -------------------------------
__device__ inline void async_b128(unsigned lds_off, const bf16_t* gaddr) {
  asm volatile("global_load_async_to_lds_b128 %0, %1, off"
               :: "v"(lds_off), "v"(gaddr) : "memory");
}
template <int N>
__device__ inline void wait_asynccnt() {
  asm volatile("s_wait_asynccnt %0" :: "i"(N));
}

// ---- CDNA5 Tensor Data Mover: one 2-D tile (rows x 32 bf16) -> LDS ---------
// D# per CDNA5 ISA. Padding: every 16 DWORDs (64 B row)
// insert 4 DWORDs (16 B) -> LDS row stride 80 B (matches LPAD below).
__device__ inline void tdm_load_tile_a(bf16_t* lds_dst, const bf16_t* gsrc, int ldk) {
  const unsigned           lds = (unsigned)(size_t)lds_dst;       // LDS byte offset
  const unsigned long long ga  = (unsigned long long)(size_t)gsrc;
  u32x4 g0;
  g0[0] = 1u;                                              // count=1, user mode
  g0[1] = lds;                                             // lds_addr [63:32]
  g0[2] = (unsigned)ga;                                    // global_addr [95:64]
  g0[3] = (unsigned)((ga >> 32) & 0x01ffffffu) | (2u << 30);  // addr[56:32] | type=2
  i32x8 g1;
  g1[0] = (1 << 16) | (1 << 20) | (3 << 22) | (3 << 25);   // data=2B, pad_en, int=16dw, amt=4dw
  g1[1] = (int)(((unsigned)ldk & 0xffffu) << 16);          // tensor_dim0 lo -> [63:48]
  g1[2] = (int)(((unsigned)ldk >> 16) | (128u << 16));     // tensor_dim0 hi | tensor_dim1=128
  g1[3] = (int)(32u << 16);                                // tensor_dim1 hi=0 | tile_dim0=32
  g1[4] = (int)128;                                        // tile_dim1=128, tile_dim2=0
  g1[5] = (int)(unsigned)ldk;                              // tensor_dim0_stride lo (elems)
  g1[6] = 0;                                               // stride0 hi | dim1_stride lo
  g1[7] = 0;
  const i32x4 z4 = {0, 0, 0, 0};                           // groups 2/3 unused (2-D)
  const i32x8 z8 = {0, 0, 0, 0, 0, 0, 0, 0};               // 6-arg toolchain extra group
  __builtin_amdgcn_tensor_load_to_lds(g0, g1, z4, z4, z8, 0);
}

// ---------------------------------------------------------------------------
// fp32 -> bf16 conversion, 4 elements/thread
// ---------------------------------------------------------------------------
__global__ __launch_bounds__(256)
void cvt_f32_to_bf16_kernel(const float* __restrict__ in, bf16_t* __restrict__ out, int n4) {
  int i = blockIdx.x * 256 + threadIdx.x;
  if (i < n4) {
    const float4 f = ((const float4*)in)[i];
    bf16_t* o = out + (size_t)i * 4;
    o[0] = (bf16_t)f.x; o[1] = (bf16_t)f.y; o[2] = (bf16_t)f.z; o[3] = (bf16_t)f.w;
  }
}

// ---------------------------------------------------------------------------
// C[M,N] = A[M,K] * W[N,K]^T ; A,W bf16 row-major (K contiguous), C f32 or bf16.
// Block: 256 threads = 8 waves, 128x128 tile; wave tile 32x64 (2x4 WMMA tiles).
// A K-slab staged by TDM (wave 0, TENSORcnt), B K-slab by async loads
// (all threads, ASYNCcnt); both double-buffered.
// ---------------------------------------------------------------------------
constexpr int LPAD = 40;   // padded LDS row length in bf16 elems (data in [0,32))

template <int OUT_F32>
__global__ __launch_bounds__(256)
void gemm_xt_kernel(const bf16_t* __restrict__ A, const bf16_t* __restrict__ W,
                    void* __restrict__ Cout, int Mm, int Nn, int Kk) {
  __shared__ __align__(16) bf16_t shA[2][128][LPAD];
  __shared__ __align__(16) bf16_t shB[2][128][LPAD];

  const int tid  = threadIdx.x;
  const int lane = tid & 31;
  const int wv   = tid >> 5;
  const int l15  = lane & 15, hh = lane >> 4;
  const int mblk = blockIdx.y * 128, nblk = blockIdx.x * 128;
  const int wm   = (wv >> 1) * 32;   // wave row offset within block: 0/32/64/96
  const int wn   = (wv & 1) * 64;    // wave col offset within block: 0/64

  const bf16_t* Ab = A + (size_t)mblk * Kk;
  const bf16_t* Wb = W + (size_t)nblk * Kk;

  auto issue_slab = [&](int nb, int k0) {
    // B tile: 2 x 16B async copies per thread (512 chunks / 256 threads)
    const unsigned lb = (unsigned)(size_t)&shB[nb][0][0];
#pragma unroll
    for (int i = 0; i < 2; ++i) {
      const int c = tid + 256 * i;
      const int row = c >> 2, seg = c & 3;
      async_b128(lb + (unsigned)(row * (LPAD * 2) + seg * 16),
                 Wb + (size_t)row * Kk + k0 + seg * 8);
    }
    // A tile: one TDM descriptor moves the whole 128x32 slab (wave 0 issues)
    if (wv == 0) tdm_load_tile_a(&shA[nb][0][0], Ab + k0, Kk);
  };

  v8f acc[2][4];
#pragma unroll
  for (int i = 0; i < 2; ++i)
#pragma unroll
    for (int j = 0; j < 4; ++j) acc[i][j] = vzero8();

  const int nk = Kk / 32;
  issue_slab(0, 0);
  for (int ks = 0; ks < nk; ++ks) {
    const int buf = ks & 1;
    if (ks + 1 < nk) {
      issue_slab(buf ^ 1, (ks + 1) * 32);
      if (wv == 0) __builtin_amdgcn_s_wait_tensorcnt(1);  // slab ks TDM done
      wait_asynccnt<2>();                                 // slab ks async done
    } else {
      if (wv == 0) __builtin_amdgcn_s_wait_tensorcnt(0);
      wait_asynccnt<0>();
    }
    __syncthreads();        // publish LDS slab across all 8 waves

    Frag a[2], bm[4];
#pragma unroll
    for (int i = 0; i < 2; ++i) {   // A frag: row fixed per lane, K runs hh*8 / 16+hh*8
      const bf16_t* pa = &shA[buf][wm + 16 * i + l15][hh * 8];
      a[i].q[0] = *(const u32x4*)pa;
      a[i].q[1] = *(const u32x4*)(pa + 16);
    }
#pragma unroll
    for (int j = 0; j < 4; ++j) {   // B frag: col fixed per lane, K runs hh*8 / 16+hh*8
      const bf16_t* pb = &shB[buf][wn + 16 * j + l15][hh * 8];
      bm[j].q[0] = *(const u32x4*)pb;
      bm[j].q[1] = *(const u32x4*)(pb + 16);
    }
#pragma unroll
    for (int i = 0; i < 2; ++i)
#pragma unroll
      for (int j = 0; j < 4; ++j)
        acc[i][j] = WMMA_BF16(a[i].v, bm[j].v, acc[i][j]);

    __syncthreads();        // all reads done before slab gets overwritten
  }

#pragma unroll
  for (int i = 0; i < 2; ++i)
#pragma unroll
    for (int j = 0; j < 4; ++j)
#pragma unroll
      for (int r = 0; r < 8; ++r) {
        const int row = mblk + wm + 16 * i + hh * 8 + r;
        const int col = nblk + wn + 16 * j + l15;
        if (OUT_F32) ((float*)Cout)[(size_t)row * Nn + col] = acc[i][j][r];
        else         ((bf16_t*)Cout)[(size_t)row * Nn + col] = (bf16_t)acc[i][j][r];
      }
}

// ---------------------------------------------------------------------------
// RoPE on Q,K (+fold softmax scale into Q), relayout:
//   qlin/klin/vlin: (B,S,DIM) bf16  ->  qh,kh: (B,H,S,Dh) bf16 ; vt: (B,H,Dh,S) bf16
// ---------------------------------------------------------------------------
__global__ __launch_bounds__(256)
void prep_rope_kernel(const bf16_t* __restrict__ qlin, const bf16_t* __restrict__ klin,
                      const bf16_t* __restrict__ vlin,
                      bf16_t* __restrict__ qh, bf16_t* __restrict__ kh,
                      bf16_t* __restrict__ vt) {
  const unsigned tid = blockIdx.x * 256 + threadIdx.x;   // B*S*H*32 total
  const int d = tid & 31;
  const int h = (tid >> 5) & 15;
  const int s = (tid >> 9) & 2047;
  const int b = tid >> 20;

  const size_t src = ((size_t)(b * Ss + s)) * DIMc + h * Dh;
  const float q1 = (float)qlin[src + d], q2 = (float)qlin[src + d + 32];
  const float k1 = (float)klin[src + d], k2 = (float)klin[src + d + 32];
  const float v1 = (float)vlin[src + d], v2 = (float)vlin[src + d + 32];

  const float freq = __expf(-(float)(2 * d) * (9.210340371976184f / 64.0f));
  const float ang  = (float)s * freq;
  const float c = __cosf(ang), sn = __sinf(ang);

  const float qo1 = (q1 * c - q2 * sn) * 0.125f;
  const float qo2 = (q2 * c + q1 * sn) * 0.125f;
  const float ko1 =  k1 * c - k2 * sn;
  const float ko2 =  k2 * c + k1 * sn;

  const size_t dsth = (((size_t)(b * Hh + h)) * Ss + s) * Dh;
  qh[dsth + d]      = (bf16_t)qo1;
  qh[dsth + d + 32] = (bf16_t)qo2;
  kh[dsth + d]      = (bf16_t)ko1;
  kh[dsth + d + 32] = (bf16_t)ko2;

  const size_t vdst = (((size_t)(b * Hh + h)) * Dh + d) * Ss + s;
  vt[vdst]                   = (bf16_t)v1;
  vt[vdst + (size_t)32 * Ss] = (bf16_t)v2;
}

// ---------------------------------------------------------------------------
// Causal flash attention. Block = 256 thr = 8 waves; each wave: 16 q-rows,
// full Dh=64. K-loop over 32-column blocks:
//   scores (4 WMMA) -> online softmax (shuffle reductions in wave halves)
//   -> P via LDS into A-layout -> PV (4 WMMA).
// ---------------------------------------------------------------------------
__global__ __launch_bounds__(256)
void flash_attn_kernel(const bf16_t* __restrict__ qh, const bf16_t* __restrict__ kh,
                       const bf16_t* __restrict__ vt, bf16_t* __restrict__ ao) {
  __shared__ __align__(16) bf16_t pSh[8][16][32];   // per-wave P staging (8 KB)

  const int lane = threadIdx.x & 31;
  const int wv   = threadIdx.x >> 5;
  const int l15  = lane & 15;
  const int hh   = lane >> 4;
  const int hh8  = hh * 8;
  const int h = blockIdx.y, b = blockIdx.z;
  const int qbase = blockIdx.x * 128 + wv * 16;

  const bf16_t* qp = qh + ((size_t)(b * Hh + h)) * Ss * Dh;   // (S, Dh)
  const bf16_t* kp = kh + ((size_t)(b * Hh + h)) * Ss * Dh;   // (S, Dh)
  const bf16_t* vp = vt + ((size_t)(b * Hh + h)) * Dh * Ss;   // (Dh, S)

  Frag aq[2];
#pragma unroll
  for (int c = 0; c < 2; ++c) {
    const bf16_t* p = qp + (size_t)(qbase + l15) * Dh + c * 32 + hh8;
    aq[c].q[0] = *(const u32x4*)p;
    aq[c].q[1] = *(const u32x4*)(p + 16);
  }

  float mreg[8], lreg[8];
  v8f o[4];
#pragma unroll
  for (int r = 0; r < 8; ++r) { mreg[r] = -1e30f; lreg[r] = 0.0f; }
#pragma unroll
  for (int t = 0; t < 4; ++t) o[t] = vzero8();

  const int kblocks = (qbase + 16 + 31) / 32;
  for (int kb = 0; kb < kblocks; ++kb) {
    const int kcol0 = kb * 32;

    v8f s0 = vzero8(), s1 = vzero8();
#pragma unroll
    for (int g = 0; g < 2; ++g) {
#pragma unroll
      for (int c = 0; c < 2; ++c) {
        Frag bk;
        const bf16_t* p = kp + (size_t)(kcol0 + g * 16 + l15) * Dh + c * 32 + hh8;
        bk.q[0] = *(const u32x4*)p;
        bk.q[1] = *(const u32x4*)(p + 16);
        if (g == 0) s0 = WMMA_BF16(aq[c].v, bk.v, s0);
        else        s1 = WMMA_BF16(aq[c].v, bk.v, s1);
      }
    }

    const int col0 = kcol0 + l15, col1 = col0 + 16;
#pragma unroll
    for (int r = 0; r < 8; ++r) {
      const int row = qbase + hh8 + r;
      float v0 = (col0 <= row) ? s0[r] : -1e30f;
      float v1 = (col1 <= row) ? s1[r] : -1e30f;
      float rm = fmaxf(v0, v1);
#pragma unroll
      for (int msk = 8; msk >= 1; msk >>= 1) rm = fmaxf(rm, __shfl_xor(rm, msk, 32));
      const float nm    = fmaxf(mreg[r], rm);
      const float alpha = __expf(mreg[r] - nm);
      const float p0 = __expf(v0 - nm);
      const float p1 = __expf(v1 - nm);
      float ps = p0 + p1;
#pragma unroll
      for (int msk = 8; msk >= 1; msk >>= 1) ps += __shfl_xor(ps, msk, 32);
      lreg[r] = lreg[r] * alpha + ps;
      mreg[r] = nm;
      o[0][r] *= alpha; o[1][r] *= alpha; o[2][r] *= alpha; o[3][r] *= alpha;
      pSh[wv][hh8 + r][l15]      = (bf16_t)p0;
      pSh[wv][hh8 + r][l15 + 16] = (bf16_t)p1;
    }

    __builtin_amdgcn_wave_barrier();
    Frag pa;
    pa.q[0] = *(const u32x4*)&pSh[wv][l15][hh8];
    pa.q[1] = *(const u32x4*)&pSh[wv][l15][16 + hh8];
    __builtin_amdgcn_wave_barrier();

#pragma unroll
    for (int t = 0; t < 4; ++t) {
      Frag bv;
      const bf16_t* p = vp + (size_t)(16 * t + l15) * Ss + kcol0 + hh8;
      bv.q[0] = *(const u32x4*)p;
      bv.q[1] = *(const u32x4*)(p + 16);
      o[t] = WMMA_BF16(pa.v, bv.v, o[t]);
    }
  }

#pragma unroll
  for (int r = 0; r < 8; ++r) {
    const float inv = 1.0f / lreg[r];
    const int srow = qbase + hh8 + r;
    const size_t base = ((size_t)b * Ss + srow) * DIMc + h * Dh;
#pragma unroll
    for (int t = 0; t < 4; ++t)
      ao[base + 16 * t + l15] = (bf16_t)(o[t][r] * inv);
  }
}

// ---------------------------------------------------------------------------
// Host launcher
// ---------------------------------------------------------------------------
extern "C" void kernel_launch(void* const* d_in, const int* in_sizes, int n_in,
                              void* d_out, int out_size, void* d_ws, size_t ws_size,
                              hipStream_t stream) {
  (void)in_sizes; (void)n_in; (void)out_size; (void)ws_size;
  const float* x  = (const float*)d_in[0];
  const float* Wq = (const float*)d_in[1];
  const float* Wk = (const float*)d_in[2];
  const float* Wv = (const float*)d_in[3];
  const float* Wo = (const float*)d_in[4];
  float* out = (float*)d_out;

  char* ws = (char*)d_ws;
  const size_t MB = 1ull << 20;
  bf16_t* xb   = (bf16_t*)(ws +   0 * MB);   // 16 MB  x bf16
  bf16_t* wqb  = (bf16_t*)(ws +  16 * MB);
  bf16_t* wkb  = (bf16_t*)(ws +  18 * MB);
  bf16_t* wvb  = (bf16_t*)(ws +  20 * MB);
  bf16_t* wob  = (bf16_t*)(ws +  22 * MB);
  bf16_t* qlin = (bf16_t*)(ws +  24 * MB);   // (B,S,DIM)
  bf16_t* klin = (bf16_t*)(ws +  40 * MB);
  bf16_t* vlin = (bf16_t*)(ws +  56 * MB);
  bf16_t* qhd  = (bf16_t*)(ws +  72 * MB);   // (B,H,S,Dh)
  bf16_t* khd  = (bf16_t*)(ws +  88 * MB);
  bf16_t* vtd  = (bf16_t*)(ws + 104 * MB);   // (B,H,Dh,S)
  bf16_t* ao   = (bf16_t*)(ws + 120 * MB);   // (B,S,DIM)

  cvt_f32_to_bf16_kernel<<<(Mrows * DIMc / 4) / 256, 256, 0, stream>>>(x,  xb,  Mrows * DIMc / 4);
  cvt_f32_to_bf16_kernel<<<(DIMc * DIMc / 4) / 256, 256, 0, stream>>>(Wq, wqb, DIMc * DIMc / 4);
  cvt_f32_to_bf16_kernel<<<(DIMc * DIMc / 4) / 256, 256, 0, stream>>>(Wk, wkb, DIMc * DIMc / 4);
  cvt_f32_to_bf16_kernel<<<(DIMc * DIMc / 4) / 256, 256, 0, stream>>>(Wv, wvb, DIMc * DIMc / 4);
  cvt_f32_to_bf16_kernel<<<(DIMc * DIMc / 4) / 256, 256, 0, stream>>>(Wo, wob, DIMc * DIMc / 4);

  dim3 ggrid(DIMc / 128, Mrows / 128);
  gemm_xt_kernel<0><<<ggrid, 256, 0, stream>>>(xb, wqb, (void*)qlin, Mrows, DIMc, DIMc);
  gemm_xt_kernel<0><<<ggrid, 256, 0, stream>>>(xb, wkb, (void*)klin, Mrows, DIMc, DIMc);
  gemm_xt_kernel<0><<<ggrid, 256, 0, stream>>>(xb, wvb, (void*)vlin, Mrows, DIMc, DIMc);

  prep_rope_kernel<<<(Bb * Ss * Hh * 32) / 256, 256, 0, stream>>>(qlin, klin, vlin,
                                                                  qhd, khd, vtd);

  flash_attn_kernel<<<dim3(Ss / 128, Hh, Bb), 256, 0, stream>>>(qhd, khd, vtd, ao);

  gemm_xt_kernel<1><<<ggrid, 256, 0, stream>>>(ao, wob, (void*)out, Mrows, DIMc, DIMc);
}
